// SimplifiedMambaBlock_22016002359936
// MI455X (gfx1250) — compile-verified, matrix operand order checked by reading, not run on hardware
//
#include <hip/hip_runtime.h>

typedef __attribute__((ext_vector_type(16))) __bf16 v16bf;
typedef __attribute__((ext_vector_type(8)))  __bf16 v8bf;
typedef __attribute__((ext_vector_type(8)))  float  v8f;

#define TILE_K 32
#define LDS_PAD 8
#define LDS_LD (TILE_K + LDS_PAD)   // 40 halfwords/row -> 80B stride, spreads banks

__device__ __forceinline__ float silu_f(float v) {
    return v / (1.f + __expf(-v));
}
__device__ __forceinline__ float softplus_f(float v) {
    return (v > 20.f) ? v : log1pf(__expf(v));
}

// gfx1250 async copy: per-lane 16B global -> LDS, tracked by ASYNCcnt (no VGPR staging)
__device__ __forceinline__ void async_copy_b128(unsigned lds_addr, const void* gptr) {
    asm volatile("global_load_async_to_lds_b128 %0, %1, off"
                 :: "v"(lds_addr), "v"((unsigned long long)gptr)
                 : "memory");
}
__device__ __forceinline__ unsigned lds_off(const void* p) {
    // generic pointers to LDS carry the LDS byte offset in addr[31:0] (ISA §10.2)
    return (unsigned)(unsigned long long)p;
}

__device__ __forceinline__ v16bf load_frag(const __bf16 (*tile)[LDS_LD], int row, int koff) {
    v8bf lo = *reinterpret_cast<const v8bf*>(&tile[row][koff]);
    v8bf hi = *reinterpret_cast<const v8bf*>(&tile[row][koff + 16]);
    return __builtin_shufflevector(lo, hi, 0,1,2,3,4,5,6,7,8,9,10,11,12,13,14,15);
}
__device__ __forceinline__ v8f wmma_bf16(v16bf a, v16bf b, v8f c) {
    return __builtin_amdgcn_wmma_f32_16x16x32_bf16(false, a, false, b, (short)0, c,
                                                   false, false);
}

// ---------------- fp32 -> bf16 conversion (weights, one pass, L2-resident) ------------
__global__ void __launch_bounds__(256) cvt_bf16_kernel(const float* __restrict__ src,
                                                       __bf16* __restrict__ dst, int n) {
    int i = blockIdx.x * 256 + threadIdx.x;
    if (i < n) dst[i] = (__bf16)src[i];
}

// ---------------- RMSNorm over dim=512, output bf16 ----------------
__global__ void __launch_bounds__(256) rmsnorm_kernel(const float* __restrict__ x,
                                                      const float* __restrict__ w,
                                                      __bf16* __restrict__ xn) {
    const int row = blockIdx.x;
    const int tid = threadIdx.x;
    const float* xr = x + (long)row * 512;
    float v0 = xr[tid];
    float v1 = xr[tid + 256];
    __shared__ float red[256];
    red[tid] = v0 * v0 + v1 * v1;
    __syncthreads();
    for (int s = 128; s > 0; s >>= 1) {
        if (tid < s) red[tid] += red[tid + s];
        __syncthreads();
    }
    const float scale = rsqrtf(red[0] * (1.f / 512.f) + 1e-6f);
    __bf16* xo = xn + (long)row * 512;
    xo[tid]       = (__bf16)(v0 * scale * w[tid]);
    xo[tid + 256] = (__bf16)(v1 * scale * w[tid + 256]);
}

// ---------------- WMMA GEMM: out[m,n] = epilogue( sum_k A[m,k] * W[n,k] ) -------------
// Block tile 128(M) x 256(N), 8 waves (2M x 4N); each wave owns a 64x64 output
// (4 A-frags x 4 B-frags -> 16 WMMAs per K-step, 16 ds_read_b128 -> 1.0 b128/wmma).
// Double-buffered LDS tiles filled with GLOBAL_LOAD_ASYNC_TO_LDS_B128 (6 per thread
// per K-step), overlapped with compute via s_wait_asynccnt; last iteration peeled so
// the steady-state loop is branch-free.
// MODE 0: silu -> bf16.  MODE 1: fp32, softplus on cols < 16 (dt).  MODE 2: fp32 + residual.
template<int MODE>
__global__ void __launch_bounds__(256) gemm_wmma_kernel(
        const __bf16* __restrict__ A, long lda,
        const __bf16* __restrict__ W, int N, int K,
        float* __restrict__ outF, __bf16* __restrict__ outB, long ldo,
        const float* __restrict__ resid)
{
    constexpr int BM = 128, BN = 256;
    __shared__ alignas(16) __bf16 sA[2][BM][LDS_LD];   // 2 x 128 x 40 x 2B = 20480 B
    __shared__ alignas(16) __bf16 sB[2][BN][LDS_LD];   // 2 x 256 x 40 x 2B = 40960 B

    const int tid   = threadIdx.x;
    const int lane  = tid & 31;
    const int wave  = tid >> 5;
    const int waveM = wave >> 2;   // 0..1  (64 rows each)
    const int waveN = wave & 3;    // 0..3  (64 cols each)
    const long m0 = (long)blockIdx.x * BM;
    const long n0 = (long)blockIdx.y * BN;

    // staging: chunk = 8 halfwords (16B). A: 128x4=512 chunks (2/thread);
    // B: 256x4=1024 chunks (4/thread). Thread t covers rows sr+64*j, col chunk sc.
    const int sr = tid >> 2;          // 0..63
    const int sc = (tid & 3) * 8;     // 0,8,16,24

    const __bf16* gA[2];
    const __bf16* gB[4];
    unsigned lA[2][2], lB[2][4];
    #pragma unroll
    for (int j = 0; j < 2; ++j) {
        gA[j] = A + (m0 + sr + 64 * j) * lda + sc;
        lA[0][j] = lds_off(&sA[0][sr + 64 * j][sc]);
        lA[1][j] = lds_off(&sA[1][sr + 64 * j][sc]);
    }
    #pragma unroll
    for (int j = 0; j < 4; ++j) {
        long r = n0 + sr + 64 * j;
        if (r >= N) r = N - 1;        // clamp tail rows (cols >= N never stored)
        gB[j] = W + r * (long)K + sc;
        lB[0][j] = lds_off(&sB[0][sr + 64 * j][sc]);
        lB[1][j] = lds_off(&sB[1][sr + 64 * j][sc]);
    }

    // fragment gather pattern (ISA 16-bit layout):
    // lane<16 : row = lane,    K = [0..7] and [16..23]
    // lane>=16: row = lane-16, K = [8..15] and [24..31]
    const int lr   = lane & 15;
    const int koff = (lane >> 4) * 8;

    v8f acc[4][4] = {};

    auto prefetch = [&](long kofs, int buf) {
        #pragma unroll
        for (int j = 0; j < 2; ++j) async_copy_b128(lA[buf][j], gA[j] + kofs);
        #pragma unroll
        for (int j = 0; j < 4; ++j) async_copy_b128(lB[buf][j], gB[j] + kofs);
    };
    auto compute_tile = [&](int cur) {
        v16bf af[4], bf[4];
        #pragma unroll
        for (int f = 0; f < 4; ++f) {
            af[f] = load_frag(sA[cur], waveM * 64 + f * 16 + lr, koff);
            bf[f] = load_frag(sB[cur], waveN * 64 + f * 16 + lr, koff);
        }
        #pragma unroll
        for (int fm = 0; fm < 4; ++fm)
            #pragma unroll
            for (int fn = 0; fn < 4; ++fn)
                acc[fm][fn] = wmma_bf16(af[fm], bf[fn], acc[fm][fn]);
    };

    const int T = K / TILE_K;          // >= 2 for all call sites

    prefetch(0, 0);
    for (int i = 0; i < T - 1; ++i) {  // steady state: branch-free prefetch + compute
        const int cur = i & 1;
        prefetch((long)(i + 1) * TILE_K, cur ^ 1);
        // tile i's 6 loads complete; tile i+1's 6 stay in flight (overlap with WMMA)
        asm volatile("s_wait_asynccnt 0x6" ::: "memory");
        __syncthreads();
        compute_tile(cur);
        __syncthreads();               // all reads of buf cur done before refill
    }
    asm volatile("s_wait_asynccnt 0x0" ::: "memory");
    __syncthreads();
    compute_tile((T - 1) & 1);

    // D layout: VGPR g holds (M = g, lanes 0-15) / (M = 8+g, lanes 16-31), N = lane&15
    const long mB = m0 + waveM * 64 + ((lane >> 4) << 3);
    const long nB = n0 + waveN * 64 + (lane & 15);

    #pragma unroll
    for (int fn = 0; fn < 4; ++fn) {
        const long n = nB + fn * 16;
        if (n >= N) continue;
        #pragma unroll
        for (int fm = 0; fm < 4; ++fm) {
            #pragma unroll
            for (int g = 0; g < 8; ++g) {
                const long m = mB + fm * 16 + g;
                const float v = acc[fm][fn][g];
                if constexpr (MODE == 0) {
                    outB[m * ldo + n] = (__bf16)silu_f(v);
                } else if constexpr (MODE == 1) {
                    outF[m * ldo + n] = (n < 16) ? softplus_f(v) : v;  // fused dt softplus
                } else {
                    outF[m * ldo + n] = v + resid[m * ldo + n];
                }
            }
        }
    }
}

// ---------------- sequential SSM scan over L + gating with z ----------------
__global__ void __launch_bounds__(256) scan_kernel(const float* __restrict__ ssm,
                                                   const float* __restrict__ A_log,
                                                   const __bf16* __restrict__ xz,
                                                   __bf16* __restrict__ y,
                                                   int L) {
    const int e = blockIdx.x * 256 + threadIdx.x;  // 0..1023
    const int b = blockIdx.y;
    float a[16], h[16];
    #pragma unroll
    for (int j = 0; j < 16; ++j) {
        a[j] = -__expf(A_log[(long)e * 16 + j]);
        h[j] = 0.f;
    }
    __shared__ float sdt[16];
    for (int t = 0; t < L; ++t) {
        const long rb = (long)b * L + t;
        if (threadIdx.x < 16) sdt[threadIdx.x] = ssm[rb * 1056 + threadIdx.x];
        __syncthreads();
        float s = 0.f;
        #pragma unroll
        for (int j = 0; j < 16; ++j) {
            h[j] *= __expf(sdt[j] * a[j]);   // A_t = exp(dt * A); h = A_t * h
            s += h[j];
        }
        const float Ct = ssm[rb * 1056 + 32 + e];
        const float zt = (float)xz[rb * 2048 + 1024 + e];   // silu(z) already applied
        y[rb * 1024 + e] = (__bf16)(s * Ct * zt);
        __syncthreads();
    }
}

// ---------------- host-side orchestration ----------------
extern "C" void kernel_launch(void* const* d_in, const int* in_sizes, int n_in,
                              void* d_out, int out_size, void* d_ws, size_t ws_size,
                              hipStream_t stream) {
    const float* x      = (const float*)d_in[0];  // (8,2048,512)
    const float* norm_w = (const float*)d_in[1];  // (512,)
    const float* in_w   = (const float*)d_in[2];  // (2048,512)
    const float* xp_w   = (const float*)d_in[3];  // (1056,1024)
    const float* A_log  = (const float*)d_in[4];  // (1024,16)
    const float* out_w  = (const float*)d_in[5];  // (512,1024)
    float* out = (float*)d_out;

    const int B = 8, L = 2048;
    const long M = (long)B * L;   // 16384 tokens

    // workspace layout (bytes)
    char* ws = (char*)d_ws;
    size_t off = 0;
    __bf16* w_in_bf  = (__bf16*)(ws + off); off += (size_t)2048 * 512  * 2;
    __bf16* w_xp_bf  = (__bf16*)(ws + off); off += (size_t)1056 * 1024 * 2;
    __bf16* w_out_bf = (__bf16*)(ws + off); off += (size_t)512  * 1024 * 2;
    __bf16* xn       = (__bf16*)(ws + off); off += (size_t)M * 512  * 2;
    __bf16* xz       = (__bf16*)(ws + off); off += (size_t)M * 2048 * 2;
    float*  ssm      = (float*) (ws + off); off += (size_t)M * 1056 * 4;
    __bf16* yb       = (__bf16*)(ws + off); off += (size_t)M * 1024 * 2;

    // weight conversions (weights stay in L2: 192 MB global L2)
    cvt_bf16_kernel<<<(2048 * 512  + 255) / 256, 256, 0, stream>>>(in_w,  w_in_bf,  2048 * 512);
    cvt_bf16_kernel<<<(1056 * 1024 + 255) / 256, 256, 0, stream>>>(xp_w,  w_xp_bf,  1056 * 1024);
    cvt_bf16_kernel<<<(512  * 1024 + 255) / 256, 256, 0, stream>>>(out_w, w_out_bf, 512 * 1024);

    // RMSNorm -> bf16 activations
    rmsnorm_kernel<<<(unsigned)M, 256, 0, stream>>>(x, norm_w, xn);

    // in_proj + SiLU: [M,512] x [2048,512]^T -> xz bf16 [M,2048]
    {
        dim3 g((unsigned)(M / 128), 2048 / 256);
        gemm_wmma_kernel<0><<<g, 256, 0, stream>>>(xn, 512, w_in_bf, 2048, 512,
                                                   nullptr, xz, 2048, nullptr);
    }
    // x_proj (+ fused dt softplus): xq = xz[:, :1024] (lda 2048) x [1056,1024]^T -> ssm fp32
    {
        dim3 g((unsigned)(M / 128), (1056 + 255) / 256);
        gemm_wmma_kernel<1><<<g, 256, 0, stream>>>(xz, 2048, w_xp_bf, 1056, 1024,
                                                   ssm, nullptr, 1056, nullptr);
    }
    // sequential scan + gate with silu(z) -> y bf16 [M,1024]
    {
        dim3 g(1024 / 256, B);
        scan_kernel<<<g, 256, 0, stream>>>(ssm, A_log, xz, yb, L);
    }
    // out_proj + residual: [M,1024] x [512,1024]^T + x -> out fp32 [M,512]
    {
        dim3 g((unsigned)(M / 128), 512 / 256);
        gemm_wmma_kernel<2><<<g, 256, 0, stream>>>(yb, 1024, w_out_bf, 512, 1024,
                                                   out, nullptr, 512, x);
    }
}
